// GraphNetwork_52699248722537
// MI455X (gfx1250) — compile-verified, matrix operand order checked by reading, not run on hardware
//
#include <hip/hip_runtime.h>
#include <hip/hip_bf16.h>
#include <stdint.h>

// ---------------------------------------------------------------------------
// Types for CDNA5 WMMA (gfx1250, wave32)
// ---------------------------------------------------------------------------
typedef __attribute__((ext_vector_type(16))) __bf16         v16bf;
typedef __attribute__((ext_vector_type(8)))  float          v8f;
typedef __attribute__((ext_vector_type(8)))  unsigned short u16x8;

union BFrag {
    u16x8 h[2];
    v16bf v;
};

static __device__ __forceinline__ unsigned short f2bf(float f) {
    unsigned int u = __float_as_uint(f);
    // round-to-nearest-even bf16 truncation
    unsigned int r = (u + 0x7FFFu + ((u >> 16) & 1u)) >> 16;
    return (unsigned short)r;
}

static __device__ __forceinline__ unsigned int pack2bf(float lo, float hi) {
    return (unsigned int)f2bf(lo) | ((unsigned int)f2bf(hi) << 16);
}

#define NNODES   50000
#define NEDGES   800000
#define FIN0     24
#define HDIM     128
#define NGRAPHS  100
#define BN_EPS   1e-5f

// ---------------------------------------------------------------------------
// Weight prep: W [Kin][H] f32 row-major  ->  Wt [H][Kpad] bf16 (zero padded)
// ---------------------------------------------------------------------------
__global__ void wprep_kernel(const float* __restrict__ W,
                             unsigned short* __restrict__ Wt,
                             int Kin, int Kpad) {
    int idx = blockIdx.x * blockDim.x + threadIdx.x;
    if (idx >= HDIM * Kpad) return;
    int n = idx / Kpad;
    int k = idx % Kpad;
    float v = (k < Kin) ? W[(size_t)k * HDIM + n] : 0.0f;
    Wt[idx] = f2bf(v);
}

// ---------------------------------------------------------------------------
// EdgeConv: out[dst] = max over edges of relu([h_dst, h_src - h_dst] @ W + b)
//   K    : padded concat width (multiple of 32); FINN: input features/node
// 256-thread block = 8 waves; each wave owns a 16-wide column tile of H=128.
// Loops over TILES 16-edge tiles; A staged in LDS as bf16, B preloaded in
// registers, bias folded into the accumulator init, segment-max via u32
// atomics on the (non-negative) f32 bit pattern.
// ---------------------------------------------------------------------------
template <int K, int FINN>
__global__ void edgeconv_kernel(const float* __restrict__ hin,
                                const int* __restrict__ esrc,
                                const int* __restrict__ edst,
                                const unsigned short* __restrict__ Wt,
                                const float* __restrict__ bias,
                                float* __restrict__ out, int E) {
    constexpr int NK    = K / 32;  // 16x16x32 wmma steps
    constexpr int TILES = 16;      // edge tiles per block
    constexpr int CH    = K / 16;  // bf16 values each thread stages (4 or 16)
    constexpr int NV    = CH / 4;  // float4 loads per thread

    __shared__ unsigned short smem[16 * K];
    __shared__ __align__(32) int sdst[16];

    const int tid  = threadIdx.x;
    const int lane = tid & 31;
    const int wave = tid >> 5;
    const int col  = lane & 15;
    const int hi   = (lane >> 4) & 1;
    const int nbase = wave * 16;

    // Preload B fragments (weights) for this wave's 16 columns.
    BFrag bfr[NK];
    const unsigned short* wp = Wt + (size_t)(nbase + col) * K + hi * 8;
#pragma unroll
    for (int kb = 0; kb < NK; ++kb) {
        bfr[kb].h[0] = *reinterpret_cast<const u16x8*>(wp + kb * 32);
        bfr[kb].h[1] = *reinterpret_cast<const u16x8*>(wp + kb * 32 + 16);
    }
    const float bval = bias[nbase + col];

    // staging geometry (same every tile)
    const int el = tid >> 4;           // edge slot 0..15
    const int k0 = (tid & 15) * CH;    // feature chunk base
    const int row = lane & 15;         // A-matrix row M

    for (int t = 0; t < TILES; ++t) {
        const int ebase = (blockIdx.x * TILES + t) * 16;
        const bool fullTile = (ebase + 16 <= E);

        __syncthreads();  // previous tile's LDS reads complete

        // ---- build A tile in LDS: 16 edges x K bf16 ----
        {
            const int e  = ebase + el;
            const bool valid = (e < E);
            const int si = valid ? esrc[e] : 0;
            const int di = valid ? edst[e] : 0;
            const float4* pd4 = reinterpret_cast<const float4*>(hin + (size_t)di * FINN);
            const float4* ps4 = reinterpret_cast<const float4*>(hin + (size_t)si * FINN);

            float f[CH];
            // chunk boundaries (FINN, 2*FINN) are multiples of CH for both layers
            if (k0 + CH <= FINN) {
#pragma unroll
                for (int v = 0; v < NV; ++v) {
                    float4 a = pd4[(k0 >> 2) + v];
                    f[4 * v + 0] = a.x; f[4 * v + 1] = a.y;
                    f[4 * v + 2] = a.z; f[4 * v + 3] = a.w;
                }
            } else if (k0 >= FINN && k0 + CH <= 2 * FINN) {
                const int b = (k0 - FINN) >> 2;
#pragma unroll
                for (int v = 0; v < NV; ++v) {
                    float4 a = ps4[b + v];
                    float4 d = pd4[b + v];
                    f[4 * v + 0] = a.x - d.x; f[4 * v + 1] = a.y - d.y;
                    f[4 * v + 2] = a.z - d.z; f[4 * v + 3] = a.w - d.w;
                }
            } else {
#pragma unroll
                for (int j = 0; j < CH; ++j) f[j] = 0.0f;
            }
            if (!valid) {
#pragma unroll
                for (int j = 0; j < CH; ++j) f[j] = 0.0f;
            }
            unsigned int* sp = reinterpret_cast<unsigned int*>(&smem[el * K + k0]);
#pragma unroll
            for (int j = 0; j < CH / 2; ++j)
                sp[j] = pack2bf(f[2 * j], f[2 * j + 1]);
            if ((tid & 15) == 0) sdst[el] = valid ? di : -1;
        }
        __syncthreads();

        // ---- WMMA over K; accumulator pre-loaded with bias ----
        v8f c;
#pragma unroll
        for (int i = 0; i < 8; ++i) c[i] = bval;

#pragma unroll
        for (int kb = 0; kb < NK; ++kb) {
            BFrag a;
            const unsigned short* ap = &smem[row * K + kb * 32 + hi * 8];
            a.h[0] = *reinterpret_cast<const u16x8*>(ap);
            a.h[1] = *reinterpret_cast<const u16x8*>(ap + 16);
            c = __builtin_amdgcn_wmma_f32_16x16x32_bf16(
                    false, a.v, false, bfr[kb].v, (short)0, c, false, false);
        }

        // ---- relu + segment-max (u32 atomic on non-negative f32 bits) ----
        const int4 sd0 = *reinterpret_cast<const int4*>(&sdst[hi * 8]);
        const int4 sd1 = *reinterpret_cast<const int4*>(&sdst[hi * 8 + 4]);
        const int dsts[8] = {sd0.x, sd0.y, sd0.z, sd0.w,
                             sd1.x, sd1.y, sd1.z, sd1.w};
        const int colOff = nbase + col;
        if (fullTile) {
#pragma unroll
            for (int r = 0; r < 8; ++r) {
                float v = c[r] > 0.0f ? c[r] : 0.0f;
                atomicMax(reinterpret_cast<unsigned int*>(
                              &out[(size_t)dsts[r] * HDIM + colOff]),
                          __float_as_uint(v));
            }
        } else {
#pragma unroll
            for (int r = 0; r < 8; ++r) {
                if (dsts[r] >= 0) {
                    float v = c[r] > 0.0f ? c[r] : 0.0f;
                    atomicMax(reinterpret_cast<unsigned int*>(
                                  &out[(size_t)dsts[r] * HDIM + colOff]),
                              __float_as_uint(v));
                }
            }
        }
    }
}

// ---------------------------------------------------------------------------
// BatchNorm partial sums: stats[0..127]=sum, stats[128..255]=sumsq
// ---------------------------------------------------------------------------
__global__ void bn_partial_kernel(const float* __restrict__ h,
                                  float* __restrict__ stats, int N) {
    const int f  = threadIdx.x;           // 128 threads = feature
    const int n0 = blockIdx.x * 256;
    float s = 0.0f, s2 = 0.0f;
    for (int i = 0; i < 256; ++i) {
        const int n = n0 + i;
        if (n < N) {
            float v = h[(size_t)n * HDIM + f];
            s  += v;
            s2 += v * v;
        }
    }
    atomicAdd(&stats[f], s);
    atomicAdd(&stats[HDIM + f], s2);
}

__global__ void bn_apply_kernel(const float* __restrict__ h,
                                float* __restrict__ out,
                                const float* __restrict__ stats,
                                const float* __restrict__ g,
                                const float* __restrict__ be,
                                int N, int do_relu) {
    const long long idx = (long long)blockIdx.x * blockDim.x + threadIdx.x;
    if (idx >= (long long)N * HDIM) return;
    const int f = (int)(idx % HDIM);
    const float invN = 1.0f / (float)N;
    const float mu   = stats[f] * invN;
    const float var  = stats[HDIM + f] * invN - mu * mu;
    float y = g[f] * (h[idx] - mu) * rsqrtf(var + BN_EPS) + be[f];
    if (do_relu) y = y > 0.0f ? y : 0.0f;
    out[idx] = y;
}

// ---------------------------------------------------------------------------
// Global mean pool: atomic adds into pooled[G][H], counts[G]
// ---------------------------------------------------------------------------
__global__ void pool_kernel(const float* __restrict__ h,
                            const int* __restrict__ batch,
                            float* __restrict__ pooled,
                            float* __restrict__ counts, int N) {
    const int f  = threadIdx.x;  // 128 threads
    const int n0 = blockIdx.x * 64;
    for (int i = 0; i < 64; ++i) {
        const int n = n0 + i;
        if (n < N) {
            const int gidx = batch[n];
            atomicAdd(&pooled[(size_t)gidx * HDIM + f], h[(size_t)n * HDIM + f]);
            if (f == 0) atomicAdd(&counts[gidx], 1.0f);
        }
    }
}

// ---------------------------------------------------------------------------
// Head: out[g] = relu(mean_pool[g] . Wl + bl)
// ---------------------------------------------------------------------------
__global__ void head_kernel(const float* __restrict__ pooled,
                            const float* __restrict__ counts,
                            const float* __restrict__ Wl,
                            const float* __restrict__ bl,
                            float* __restrict__ out) {
    __shared__ float red[128];
    const int g = blockIdx.x;
    const int t = threadIdx.x;
    const float cnt = counts[g] > 1.0f ? counts[g] : 1.0f;
    red[t] = (pooled[(size_t)g * HDIM + t] / cnt) * Wl[t];
    __syncthreads();
    for (int s = 64; s > 0; s >>= 1) {
        if (t < s) red[t] += red[t + s];
        __syncthreads();
    }
    if (t == 0) {
        float v = red[0] + bl[0];
        out[g] = v > 0.0f ? v : 0.0f;
    }
}

// ---------------------------------------------------------------------------
// Launcher
// ---------------------------------------------------------------------------
extern "C" void kernel_launch(void* const* d_in, const int* in_sizes, int n_in,
                              void* d_out, int out_size, void* d_ws, size_t ws_size,
                              hipStream_t stream) {
    const float* x   = (const float*)d_in[0];
    const int*   ei  = (const int*)d_in[1];
    const int*   bat = (const int*)d_in[2];
    const float* W1  = (const float*)d_in[3];
    const float* b1  = (const float*)d_in[4];
    const float* W2  = (const float*)d_in[5];
    const float* b2  = (const float*)d_in[6];
    const float* W3  = (const float*)d_in[7];
    const float* b3  = (const float*)d_in[8];
    const float* g1  = (const float*)d_in[9];
    const float* be1 = (const float*)d_in[10];
    const float* g2  = (const float*)d_in[11];
    const float* be2 = (const float*)d_in[12];
    const float* g3  = (const float*)d_in[13];
    const float* be3 = (const float*)d_in[14];
    const float* Wl  = (const float*)d_in[15];
    const float* bl  = (const float*)d_in[16];
    float* out = (float*)d_out;

    const int N = in_sizes[0] / FIN0;   // 50000
    const int E = in_sizes[1] / 2;      // 800000
    const int* esrc = ei;
    const int* edst = ei + E;

    // Workspace carve-up (256B aligned)
    char* ws = (char*)d_ws;
    size_t off = 0;
    auto alloc = [&](size_t bytes) {
        char* p = ws + off;
        off = (off + bytes + 255) & ~(size_t)255;
        return p;
    };
    const size_t NB = (size_t)N * HDIM * sizeof(float);
    float*          hA     = (float*)alloc(NB);
    float*          hB     = (float*)alloc(NB);
    unsigned short* Wt1    = (unsigned short*)alloc((size_t)HDIM * 64 * 2);
    unsigned short* Wt2    = (unsigned short*)alloc((size_t)HDIM * 256 * 2);
    unsigned short* Wt3    = (unsigned short*)alloc((size_t)HDIM * 256 * 2);
    float*          stats  = (float*)alloc(2 * HDIM * sizeof(float));
    float*          pooled = (float*)alloc((size_t)NGRAPHS * HDIM * sizeof(float));
    float*          counts = (float*)alloc(NGRAPHS * sizeof(float));
    (void)ws_size;

    // Weight prep (f32 -> transposed bf16, zero-padded K)
    wprep_kernel<<<(HDIM * 64  + 255) / 256, 256, 0, stream>>>(W1, Wt1, 2 * FIN0, 64);
    wprep_kernel<<<(HDIM * 256 + 255) / 256, 256, 0, stream>>>(W2, Wt2, 2 * HDIM, 256);
    wprep_kernel<<<(HDIM * 256 + 255) / 256, 256, 0, stream>>>(W3, Wt3, 2 * HDIM, 256);

    const int EPW    = 16 * 16;                // edges per block (TILES=16)
    const int egrid  = (E + EPW - 1) / EPW;
    const int bnapG  = (int)(((long long)N * HDIM + 255) / 256);
    const int bnparG = (N + 255) / 256;
    const int poolG  = (N + 63) / 64;

    // ---- Layer 1: K=64 (2*24 padded), FIN=24 ----
    hipMemsetAsync(hA, 0, NB, stream);
    edgeconv_kernel<64, FIN0><<<egrid, 256, 0, stream>>>(x, esrc, edst, Wt1, b1, hA, E);
    hipMemsetAsync(stats, 0, 2 * HDIM * sizeof(float), stream);
    bn_partial_kernel<<<bnparG, 128, 0, stream>>>(hA, stats, N);
    bn_apply_kernel<<<bnapG, 256, 0, stream>>>(hA, hB, stats, g1, be1, N, 1);

    // ---- Layer 2: K=256, FIN=128 ----
    hipMemsetAsync(hA, 0, NB, stream);
    edgeconv_kernel<256, HDIM><<<egrid, 256, 0, stream>>>(hB, esrc, edst, Wt2, b2, hA, E);
    hipMemsetAsync(stats, 0, 2 * HDIM * sizeof(float), stream);
    bn_partial_kernel<<<bnparG, 128, 0, stream>>>(hA, stats, N);
    bn_apply_kernel<<<bnapG, 256, 0, stream>>>(hA, hB, stats, g2, be2, N, 1);

    // ---- Layer 3: K=256, FIN=128, BN without relu ----
    hipMemsetAsync(hA, 0, NB, stream);
    edgeconv_kernel<256, HDIM><<<egrid, 256, 0, stream>>>(hB, esrc, edst, Wt3, b3, hA, E);
    hipMemsetAsync(stats, 0, 2 * HDIM * sizeof(float), stream);
    bn_partial_kernel<<<bnparG, 128, 0, stream>>>(hA, stats, N);
    bn_apply_kernel<<<bnapG, 256, 0, stream>>>(hA, hB, stats, g3, be3, N, 0);

    // ---- Pool + head ----
    hipMemsetAsync(pooled, 0, (size_t)NGRAPHS * HDIM * sizeof(float), stream);
    hipMemsetAsync(counts, 0, NGRAPHS * sizeof(float), stream);
    pool_kernel<<<poolG, 128, 0, stream>>>(hB, bat, pooled, counts, N);
    head_kernel<<<NGRAPHS, 128, 0, stream>>>(pooled, counts, Wl, bl, out);
}
